// RELATE_1262720385474
// MI455X (gfx1250) — compile-verified, matrix operand order checked by reading, not run on hardware
//
#include <hip/hip_runtime.h>

// ---------------------------------------------------------------------------
// Problem constants (from reference setup_inputs)
// ---------------------------------------------------------------------------
#define B_   32
#define F_   40
#define NB_  36
#define R_   512
#define H_   512
#define D_   1536          // R + 2H
#define M1_  1280          // B*F

typedef __attribute__((ext_vector_type(16))) __bf16 v16bf;
typedef __attribute__((ext_vector_type(8)))  __bf16 v8bf;
typedef __attribute__((ext_vector_type(8)))  float  v8f;

// Hardware tanh (V_TANH_F32, TRANS pipe) when the toolchain exposes it;
// fall back to OCML tanhf otherwise. Guarded so compilation never breaks.
#if defined(__has_builtin)
#  if __has_builtin(__builtin_amdgcn_tanhf)
#    define FAST_TANH(x) __builtin_amdgcn_tanhf(x)
#  elif __has_builtin(__builtin_amdgcn_tanh_f32)
#    define FAST_TANH(x) __builtin_amdgcn_tanh_f32(x)
#  else
#    define FAST_TANH(x) tanhf(x)
#  endif
#else
#  define FAST_TANH(x) tanhf(x)
#endif

__device__ __forceinline__ v8f zero_v8f() {
  v8f z;
#pragma unroll
  for (int i = 0; i < 8; ++i) z[i] = 0.f;
  return z;
}

// Load one 16x32 bf16 fragment in the CDNA5 wave32 A-layout from row-major
// memory with leading dimension `ld` (elements). The same pattern also loads
// the B operand for D = A @ W^T when W is stored row-major [n][k].
//   lane&15  -> row (M or N)
//   lane>>4  -> K half: lanes 0-15 hold K {k..k+7, k+16..k+23},
//               lanes 16-31 hold K {k+8..k+15, k+24..k+31}
__device__ __forceinline__ v16bf load_frag_bf16(const __bf16* base, int ld, int lane) {
  const __bf16* p = base + (lane & 15) * ld + ((lane >> 4) << 3);
  v8bf lo = *(const v8bf*)(p);
  v8bf hi = *(const v8bf*)(p + 16);
  v16bf r;
#pragma unroll
  for (int t = 0; t < 8; ++t) { r[t] = lo[t]; r[t + 8] = hi[t]; }
  return r;
}

__device__ __forceinline__ v8f wmma_bf16(v16bf a, v16bf b, v8f c) {
  // D(f32,16x16) = A(bf16,16x32) x B(bf16,32x16) + C, f32 accumulate
  return __builtin_amdgcn_wmma_f32_16x16x32_bf16(false, a, false, b,
                                                 (short)0, c, false, false);
}

// ---------------------------------------------------------------------------
// prep: f32->bf16 weight conversion + splice i3d_feats into feat buffers
// ---------------------------------------------------------------------------
__global__ __launch_bounds__(256) void prep_kernel(
    const float* __restrict__ Ws_f, const float* __restrict__ Wr_f,
    const float* __restrict__ i3d,
    __bf16* __restrict__ Wsf_bf, __bf16* __restrict__ Wrf_bf,
    float* __restrict__ feat_f32, __bf16* __restrict__ feat_bf)
{
  const size_t stride = (size_t)gridDim.x * blockDim.x;
  const size_t t0 = (size_t)blockIdx.x * blockDim.x + threadIdx.x;
  for (size_t t = t0; t < (size_t)H_ * R_; t += stride)
    Wsf_bf[t] = (__bf16)Ws_f[t];
  for (size_t t = t0; t < (size_t)H_ * 2 * D_; t += stride)
    Wrf_bf[t] = (__bf16)Wr_f[t];
  for (size_t t = t0; t < (size_t)M1_ * 1024; t += stride) {
    size_t n = t >> 10, h = t & 1023;
    float v = i3d[t];
    size_t o = n * D_ + R_ + h;          // feat[:, 512:1536] = i3d
    feat_f32[o] = v;
    feat_bf[o]  = (__bf16)v;
  }
}

// ---------------------------------------------------------------------------
// hproj: hs = hidden @ Ws_h.T + bs_h ; hb = hidden @ Wr_h.T + br_h + br_f
// (33 MFLOP, plain f32 — keeps full precision where it's cheap)
// ---------------------------------------------------------------------------
__global__ __launch_bounds__(256) void hproj_kernel(
    const float* __restrict__ hidden,
    const float* __restrict__ Ws_h, const float* __restrict__ bs_h,
    const float* __restrict__ Wr_h, const float* __restrict__ br_h,
    const float* __restrict__ br_f,
    float* __restrict__ hs, float* __restrict__ hb)
{
  const int idx = blockIdx.x * 256 + threadIdx.x;
  if (idx >= B_ * H_) return;
  const int b = idx >> 9, h = idx & (H_ - 1);
  const float* hv = hidden + (size_t)b * H_;
  const float* w1 = Ws_h + (size_t)h * H_;
  const float* w2 = Wr_h + (size_t)h * H_;
  float a1 = 0.f, a2 = 0.f;
#pragma unroll 4
  for (int k = 0; k < H_; ++k) {
    float x = hv[k];
    a1 += x * w1[k];
    a2 += x * w2[k];
  }
  hs[idx] = a1 + bs_h[h];
  hb[idx] = a2 + br_h[h] + br_f[h];
}

// ---------------------------------------------------------------------------
// stage1: per (b,f) object soft-attention, fully fused.
//   logits = feats(36x512) @ Ws_f^T (bf16 WMMA, f32 acc)
//   e[m]   = sum_h was[h]*tanh(logits + bs_f + hs[b])
//   alpha  = softmax_36(e);  obj_att = alpha @ feats  -> feat[:, :512]
// 1280 blocks x 256 threads (8 waves); wave w owns N-columns [64w, 64w+64).
// ---------------------------------------------------------------------------
#define LDA 520   // 36(+pad to 48) x 512 bf16 tile, stride 520 to stagger banks

__global__ __launch_bounds__(256) void stage1_kernel(
    const float* __restrict__ obj_feats, const float* __restrict__ bs_f,
    const float* __restrict__ was, const float* __restrict__ hs,
    const __bf16* __restrict__ Wsf_bf,
    float* __restrict__ feat_f32, __bf16* __restrict__ feat_bf)
{
  __shared__ __bf16 Abuf[48 * LDA];
  __shared__ float  ewave[8][48];
  __shared__ float  e_lds[48];
  __shared__ float  alpha_s[NB_];

  const int n   = blockIdx.x;            // 0..1279
  const int b   = n / F_;
  const int tid = threadIdx.x;
  const int wave = tid >> 5, lane = tid & 31;

  // Stage feats (f32 global -> bf16 LDS); zero the 12 pad rows.
  const float* src = obj_feats + (size_t)n * NB_ * R_;
  for (int idx = tid; idx < NB_ * R_; idx += 256) {
    int m = idx >> 9, k = idx & (R_ - 1);
    Abuf[m * LDA + k] = (__bf16)src[idx];
  }
  for (int idx = NB_ * R_ + tid; idx < 48 * R_; idx += 256) {
    int m = idx >> 9, k = idx & (R_ - 1);
    Abuf[m * LDA + k] = (__bf16)0.f;
  }
  __syncthreads();

  const int col0 = wave * 64;            // 4 N-tiles per wave
  v8f c[3][4];
#pragma unroll
  for (int mt = 0; mt < 3; ++mt)
#pragma unroll
    for (int nt = 0; nt < 4; ++nt) c[mt][nt] = zero_v8f();

  for (int k0 = 0; k0 < R_; k0 += 32) {
    v16bf a[3];
#pragma unroll
    for (int mt = 0; mt < 3; ++mt)
      a[mt] = load_frag_bf16(&Abuf[(mt * 16) * LDA + k0], LDA, lane);
#pragma unroll
    for (int nt = 0; nt < 4; ++nt) {
      v16bf bfr = load_frag_bf16(Wsf_bf + (size_t)(col0 + nt * 16) * R_ + k0,
                                 R_, lane);
#pragma unroll
      for (int mt = 0; mt < 3; ++mt)
        c[mt][nt] = wmma_bf16(a[mt], bfr, c[mt][nt]);
    }
  }

  // Fused epilogue: tanh, dot with `was`, deterministic shuffle-tree reduce.
  float bias[4], wv[4];
#pragma unroll
  for (int nt = 0; nt < 4; ++nt) {
    int h = col0 + nt * 16 + (lane & 15);
    bias[nt] = bs_f[h] + hs[(size_t)b * H_ + h];
    wv[nt]   = was[h];
  }
  const int rofs = (lane >> 4) << 3;     // C layout: lanes>=16 hold M = v+8
#pragma unroll
  for (int mt = 0; mt < 3; ++mt) {
#pragma unroll
    for (int v = 0; v < 8; ++v) {
      float s = 0.f;
#pragma unroll
      for (int nt = 0; nt < 4; ++nt)
        s += wv[nt] * FAST_TANH(c[mt][nt][v] + bias[nt]);
#pragma unroll
      for (int off = 1; off < 16; off <<= 1)  // reduce within each 16-lane half
        s += __shfl_xor(s, off, 32);
      if ((lane & 15) == 0)
        ewave[wave][mt * 16 + rofs + v] = s;
    }
  }
  __syncthreads();
  if (tid < 48) {
    float t = 0.f;
#pragma unroll
    for (int w = 0; w < 8; ++w) t += ewave[w][tid];
    e_lds[tid] = t;
  }
  __syncthreads();
  if (tid == 0) {                         // softmax over 36 (serial, tiny)
    float mx = -3.4028235e38f;
    for (int m = 0; m < NB_; ++m) mx = fmaxf(mx, e_lds[m]);
    float sum = 0.f;
    for (int m = 0; m < NB_; ++m) {
      float ex = __expf(e_lds[m] - mx);
      alpha_s[m] = ex; sum += ex;
    }
    float inv = 1.f / sum;
    for (int m = 0; m < NB_; ++m) alpha_s[m] *= inv;
  }
  __syncthreads();
  // obj_att = alpha @ feats -> feat[:, :512] (f32 + bf16)
  for (int h = tid; h < R_; h += 256) {
    float acc = 0.f;
#pragma unroll 4
    for (int m = 0; m < NB_; ++m)
      acc += alpha_s[m] * (float)Abuf[m * LDA + h];
    size_t o = (size_t)n * D_ + h;
    feat_f32[o] = acc;
    feat_bf[o]  = (__bf16)acc;
  }
}

// ---------------------------------------------------------------------------
// p12: [p1 | p2](1280 x 1024) = feat(1280x1536) @ [Wr1 | Wr2]^T, bf16 WMMA.
// grid (20, 8): x = 64-row M block, y = 128-col N block (1 N-tile per wave).
// ---------------------------------------------------------------------------
__global__ __launch_bounds__(256) void p12_kernel(
    const __bf16* __restrict__ feat_bf, const __bf16* __restrict__ Wrf_bf,
    float* __restrict__ p12)
{
  const int m0  = blockIdx.x * 64;
  const int tid = threadIdx.x;
  const int wave = tid >> 5, lane = tid & 31;
  const int gn = blockIdx.y * 8 + wave;  // global N-tile 0..63 (virtual 1024)
  // N-tiles 0..31 -> Wr1 = Wr_f[:, :1536]; 32..63 -> Wr2 = Wr_f[:, 1536:]
  const __bf16* Bbase = (gn < 32)
      ? (Wrf_bf + (size_t)(gn * 16) * (2 * D_))
      : (Wrf_bf + (size_t)((gn - 32) * 16) * (2 * D_) + D_);

  v8f c[4];
#pragma unroll
  for (int mt = 0; mt < 4; ++mt) c[mt] = zero_v8f();

  for (int k0 = 0; k0 < D_; k0 += 32) {
    v16bf bfr = load_frag_bf16(Bbase + k0, 2 * D_, lane);
#pragma unroll
    for (int mt = 0; mt < 4; ++mt) {
      v16bf a = load_frag_bf16(feat_bf + (size_t)(m0 + mt * 16) * D_ + k0,
                               D_, lane);
      c[mt] = wmma_bf16(a, bfr, c[mt]);
    }
  }

  const int col  = gn * 16 + (lane & 15);
  const int rofs = (lane >> 4) << 3;
#pragma unroll
  for (int mt = 0; mt < 4; ++mt)
#pragma unroll
    for (int v = 0; v < 8; ++v)
      p12[(size_t)(m0 + mt * 16 + rofs + v) * 1024 + col] = c[mt][v];
}

// ---------------------------------------------------------------------------
// e[b,i,j] = sum_h war[h] * tanh(p2[b,i,h] + p1[b,j,h] + hb[b,h])
// 1280 blocks (b,i); wave w handles j = w, w+8, ... (shuffle-tree reduce).
// ---------------------------------------------------------------------------
__global__ __launch_bounds__(256) void e_kernel(
    const float* __restrict__ p12, const float* __restrict__ hb,
    const float* __restrict__ war, float* __restrict__ e)
{
  __shared__ float base_s[H_];
  __shared__ float war_s[H_];
  const int bi = blockIdx.x;             // b*40 + i
  const int b  = bi / F_;
  const int tid = threadIdx.x;
  const float* p2 = p12 + (size_t)bi * 1024 + 512;
  for (int h = tid; h < H_; h += 256) {
    base_s[h] = p2[h] + hb[(size_t)b * H_ + h];
    war_s[h]  = war[h];
  }
  __syncthreads();
  const int wave = tid >> 5, lane = tid & 31;
  for (int j = wave; j < F_; j += 8) {
    const float* p1 = p12 + (size_t)(b * F_ + j) * 1024;
    float s = 0.f;
#pragma unroll 4
    for (int h = lane; h < H_; h += 32)
      s += war_s[h] * FAST_TANH(base_s[h] + p1[h]);
#pragma unroll
    for (int off = 16; off; off >>= 1) s += __shfl_down(s, off, 32);
    if (lane == 0) e[(size_t)bi * F_ + j] = s;
  }
}

// ---------------------------------------------------------------------------
// finalize: per b, softmax over F*F=1600 logits jointly, then
//   rel1 = (col-sums of alpha) @ feat ; rel2 = (row-sums of alpha) @ feat
// ---------------------------------------------------------------------------
__global__ __launch_bounds__(256) void finalize_kernel(
    const float* __restrict__ e, const float* __restrict__ feat_f32,
    float* __restrict__ out)
{
  __shared__ float ebuf[F_ * F_];
  __shared__ float wred[8];
  __shared__ float s1[F_], s2[F_];
  __shared__ float stat_max, stat_inv;
  const int b   = blockIdx.x;
  const int tid = threadIdx.x;
  const int wave = tid >> 5, lane = tid & 31;
  const float* eb = e + (size_t)b * F_ * F_;

  float lmax = -3.4028235e38f;
  for (int t = tid; t < F_ * F_; t += 256) {
    float v = eb[t]; ebuf[t] = v; lmax = fmaxf(lmax, v);
  }
#pragma unroll
  for (int off = 16; off; off >>= 1) lmax = fmaxf(lmax, __shfl_down(lmax, off, 32));
  if (lane == 0) wred[wave] = lmax;
  __syncthreads();
  if (tid == 0) {
    float m = wred[0];
    for (int w = 1; w < 8; ++w) m = fmaxf(m, wred[w]);
    stat_max = m;
  }
  __syncthreads();
  const float mx = stat_max;
  float lsum = 0.f;
  for (int t = tid; t < F_ * F_; t += 256) {
    float ex = __expf(ebuf[t] - mx); ebuf[t] = ex; lsum += ex;
  }
#pragma unroll
  for (int off = 16; off; off >>= 1) lsum += __shfl_down(lsum, off, 32);
  __syncthreads();
  if (lane == 0) wred[wave] = lsum;
  __syncthreads();
  if (tid == 0) {
    float s = 0.f;
    for (int w = 0; w < 8; ++w) s += wred[w];
    stat_inv = 1.f / s;
  }
  __syncthreads();
  const float inv = stat_inv;
  if (tid < F_) {                        // s1[j] = inv * sum_i alpha[i][j]
    float a = 0.f;
    for (int i = 0; i < F_; ++i) a += ebuf[i * F_ + tid];
    s1[tid] = a * inv;
  } else if (tid >= 64 && tid < 64 + F_) { // s2[i] = inv * sum_j alpha[i][j]
    const int i = tid - 64;
    float a = 0.f;
    for (int j = 0; j < F_; ++j) a += ebuf[i * F_ + j];
    s2[i] = a * inv;
  }
  __syncthreads();
  const float* fb = feat_f32 + (size_t)b * F_ * D_;
  for (int d = tid; d < D_; d += 256) {
    float r1 = 0.f, r2 = 0.f;
#pragma unroll 4
    for (int t = 0; t < F_; ++t) {
      float fv = fb[(size_t)t * D_ + d];
      r1 += s1[t] * fv;
      r2 += s2[t] * fv;
    }
    out[(size_t)b * (2 * D_) + d]      = r1;
    out[(size_t)b * (2 * D_) + D_ + d] = r2;
  }
}

// ---------------------------------------------------------------------------
// host launcher
// ---------------------------------------------------------------------------
extern "C" void kernel_launch(void* const* d_in, const int* in_sizes, int n_in,
                              void* d_out, int out_size, void* d_ws, size_t ws_size,
                              hipStream_t stream) {
  (void)in_sizes; (void)n_in; (void)out_size; (void)ws_size;
  const float* i3d    = (const float*)d_in[0];
  const float* objf   = (const float*)d_in[1];
  const float* hidden = (const float*)d_in[2];
  const float* Ws_f   = (const float*)d_in[3];
  const float* bs_f   = (const float*)d_in[4];
  const float* Ws_h   = (const float*)d_in[5];
  const float* bs_h   = (const float*)d_in[6];
  const float* was    = (const float*)d_in[7];
  const float* Wr_f   = (const float*)d_in[8];
  const float* br_f   = (const float*)d_in[9];
  const float* Wr_h   = (const float*)d_in[10];
  const float* br_h   = (const float*)d_in[11];
  const float* war    = (const float*)d_in[12];
  float* out = (float*)d_out;

  // Workspace layout (~21 MB, 256B-aligned slices)
  char* ws = (char*)d_ws;
  size_t off = 0;
  auto take = [&](size_t bytes) -> char* {
    char* p = ws + off;
    off += (bytes + 255) & ~(size_t)255;
    return p;
  };
  __bf16* Wsf_bf = (__bf16*)take((size_t)H_ * R_ * 2);
  __bf16* Wrf_bf = (__bf16*)take((size_t)H_ * 2 * D_ * 2);
  float*  feat32 = (float*) take((size_t)M1_ * D_ * 4);
  __bf16* featbf = (__bf16*)take((size_t)M1_ * D_ * 2);
  float*  hs     = (float*) take((size_t)B_ * H_ * 4);
  float*  hb     = (float*) take((size_t)B_ * H_ * 4);
  float*  p12    = (float*) take((size_t)M1_ * 1024 * 4);
  float*  ebuf   = (float*) take((size_t)B_ * F_ * F_ * 4);

  prep_kernel<<<dim3(1024), dim3(256), 0, stream>>>(
      Ws_f, Wr_f, i3d, Wsf_bf, Wrf_bf, feat32, featbf);
  hproj_kernel<<<dim3((B_ * H_ + 255) / 256), dim3(256), 0, stream>>>(
      hidden, Ws_h, bs_h, Wr_h, br_h, br_f, hs, hb);
  stage1_kernel<<<dim3(M1_), dim3(256), 0, stream>>>(
      objf, bs_f, was, hs, Wsf_bf, feat32, featbf);
  p12_kernel<<<dim3(20, 8), dim3(256), 0, stream>>>(featbf, Wrf_bf, p12);
  e_kernel<<<dim3(M1_), dim3(256), 0, stream>>>(p12, hb, war, ebuf);
  finalize_kernel<<<dim3(B_), dim3(256), 0, stream>>>(ebuf, feat32, out);
}
